// GNNEncoder_49770081026336
// MI455X (gfx1250) — matrix operand order repaired and checked by reading, not compiled, not run on hardware
//
#include <hip/hip_runtime.h>
#include <hip/hip_bf16.h>

typedef __attribute__((ext_vector_type(16))) _Float16 v16h;
typedef __attribute__((ext_vector_type(8)))  float    v8f;

#define BATCH 8
#define NPTS  2048
#define KNN   20
#define MEDGE (BATCH*NPTS*KNN)   /* 327680 */
#define BN_EPS 1e-5f
#define FINF 3.402823e38f

// ---------------------------------------------------------------------------
// WMMA fragment packing per CDNA5 ISA 7.12.2 (wave32)
// A (16x32 f16, row-major in LDS): lane 0-15 -> M=lane, K in {0..7,16..23};
//                                  lane 16-31 -> M=lane-15... (M=lane&15), K in {8..15,24..31}
// ---------------------------------------------------------------------------
__device__ __forceinline__ v16h frag_a(const _Float16* p, int stride, int lane) {
  v16h a;
  const int m  = lane & 15;
  const int kb = (lane < 16) ? 0 : 8;
#pragma unroll
  for (int r = 0; r < 8; ++r) {
    int k0 = (r < 4) ? (2 * r + kb) : (16 + 2 * (r - 4) + kb);
    a[2 * r]     = p[m * stride + k0];
    a[2 * r + 1] = p[m * stride + k0 + 1];
  }
  return a;
}
// B (32x16 f16, row-major [K][N] in LDS): lane -> N=lane&15; lane 0-15 K in {0..7,16..23},
// lane 16-31 K in {8..15,24..31}; register elements run over K.
__device__ __forceinline__ v16h frag_b(const _Float16* p, int stride, int lane) {
  v16h b;
  const int n  = lane & 15;
  const int kb = (lane < 16) ? 0 : 8;
#pragma unroll
  for (int e = 0; e < 16; ++e) {
    int k = (e < 8) ? (kb + e) : (16 + kb + (e - 8));
    b[e] = p[k * stride + n];
  }
  return b;
}
// C/D (16x16 f32): lane -> N=lane&15, vgpr v -> M = v + (lane<16?0:8)

// ---------------------------------------------------------------------------
// 1) pairwise squared distances for the raw 3-D points
// ---------------------------------------------------------------------------
__global__ void k_dist1(const float* __restrict__ x, float* __restrict__ dist) {
  long t = (long)blockIdx.x * blockDim.x + threadIdx.x;
  if (t >= (long)BATCH * NPTS * NPTS) return;
  int  j = (int)(t % NPTS);
  long r = t / NPTS;
  int  i = (int)(r % NPTS);
  int  b = (int)(r / NPTS);
  const float* xi = x + ((long)b * NPTS + i) * 3;
  const float* xj = x + ((long)b * NPTS + j) * 3;
  float d0 = xi[0] - xj[0], d1 = xi[1] - xj[1], d2 = xi[2] - xj[2];
  dist[t] = d0 * d0 + d1 * d1 + d2 * d2;
}

// ---------------------------------------------------------------------------
// 2) top-KNN smallest per row: LDS copy + 20x masked argmin (stable ties)
// ---------------------------------------------------------------------------
__global__ void k_select(const float* __restrict__ dist, int* __restrict__ idx) {
  __shared__ float sd[NPTS];
  __shared__ float rmin[128];
  __shared__ int   rloc[128];
  const long  row  = blockIdx.x;
  const float* dr  = dist + row * (long)NPTS;
  for (int j = threadIdx.x; j < NPTS; j += 128) sd[j] = dr[j];
  __syncthreads();
  for (int kk = 0; kk < KNN; ++kk) {
    float best = FINF; int bi = NPTS;
    for (int j = threadIdx.x; j < NPTS; j += 128) {
      float v = sd[j];
      if (v < best || (v == best && j < bi)) { best = v; bi = j; }
    }
    rmin[threadIdx.x] = best; rloc[threadIdx.x] = bi;
    __syncthreads();
    for (int s = 64; s > 0; s >>= 1) {
      if (threadIdx.x < s) {
        float ov = rmin[threadIdx.x + s]; int oi = rloc[threadIdx.x + s];
        if (ov < rmin[threadIdx.x] ||
            (ov == rmin[threadIdx.x] && oi < rloc[threadIdx.x])) {
          rmin[threadIdx.x] = ov; rloc[threadIdx.x] = oi;
        }
      }
      __syncthreads();
    }
    if (threadIdx.x == 0) { idx[row * KNN + kk] = rloc[0]; sd[rloc[0]] = FINF; }
    __syncthreads();
  }
}

// ---------------------------------------------------------------------------
// 3) Edge-feature GEMM: gather [x_i, x_j - x_i] -> z = ef @ W + b  (WMMA f16)
//    CIN2 = 2*D (6 or 128, K padded to mult of 32), COUT = 64 or 128
// ---------------------------------------------------------------------------
template <int CIN2, int COUT>
__global__ void k_edge_gemm(const float* __restrict__ feat, const int* __restrict__ idx,
                            const float* __restrict__ W, const float* __restrict__ bias,
                            float* __restrict__ z) {
  constexpr int D      = CIN2 / 2;
  constexpr int KPAD   = (CIN2 + 31) & ~31;
  constexpr int ROWT   = (COUT == 64) ? 2 : 1;
  constexpr int WGROWS = 16 * ROWT;
  __shared__ _Float16 Wlds[KPAD * COUT];
  __shared__ _Float16 Alds[WGROWS * KPAD];
  const int tid = threadIdx.x;

  for (int t = tid; t < KPAD * COUT; t += 256) {
    int k = t / COUT, n = t % COUT;
    Wlds[t] = (k < CIN2) ? (_Float16)W[k * COUT + n] : (_Float16)0.f;
  }
  const long row0 = (long)blockIdx.x * WGROWS;
  for (int t = tid; t < WGROWS * KPAD; t += 256) {
    int m = t / KPAD, k = t % KPAD;
    float v = 0.f;
    if (k < CIN2) {
      long e  = row0 + m;
      long p  = e / KNN;                 // b*NPTS + point
      long b  = p / NPTS;
      int  nb = idx[e];
      const float* fi = feat + p * D;
      const float* fj = feat + (b * NPTS + nb) * D;
      v = (k < D) ? fi[k] : (fj[k - D] - fi[k - D]);
    }
    Alds[t] = (_Float16)v;
  }
  __syncthreads();

  const int lane = tid & 31, wave = tid >> 5;
  const int rt = (ROWT == 2) ? (wave >> 2) : 0;
  const int ct = (ROWT == 2) ? (wave & 3) : wave;
  v8f acc = {};
#pragma unroll
  for (int kt = 0; kt < KPAD / 32; ++kt) {
    v16h af = frag_a(Alds + rt * 16 * KPAD + kt * 32, KPAD, lane);
    v16h bf = frag_b(Wlds + kt * 32 * COUT + ct * 16, COUT, lane);
    acc = __builtin_amdgcn_wmma_f32_16x16x32_f16(false, af, false, bf,
                                                 (short)0, acc, false, false);
  }
  const int  n     = ct * 16 + (lane & 15);
  const float bv   = bias[n];
  const long rbase = row0 + rt * 16 + ((lane < 16) ? 0 : 8);
#pragma unroll
  for (int v = 0; v < 8; ++v) z[(rbase + v) * COUT + n] = acc[v] + bv;
}

// ---------------------------------------------------------------------------
// 4) Activation GEMM: z = relu(bn(zprev)) @ W + b  (BN fused into A-load)
// ---------------------------------------------------------------------------
template <int CIN, int COUT>
__global__ void k_act_gemm(const float* __restrict__ zprev,
                           const float* __restrict__ mean, const float* __restrict__ rstd,
                           const float* __restrict__ g, const float* __restrict__ be,
                           const float* __restrict__ W, const float* __restrict__ bias,
                           float* __restrict__ z) {
  constexpr int ROWT   = (COUT == 64) ? 2 : 1;
  constexpr int WGROWS = 16 * ROWT;
  __shared__ _Float16 Wlds[CIN * COUT];
  __shared__ _Float16 Alds[WGROWS * CIN];
  const int tid = threadIdx.x;

  for (int t = tid; t < CIN * COUT; t += 256) {
    int k = t / COUT, n = t % COUT;
    Wlds[t] = (_Float16)W[k * COUT + n];
  }
  const long row0 = (long)blockIdx.x * WGROWS;
  for (int t = tid; t < WGROWS * CIN; t += 256) {
    int m = t / CIN, k = t % CIN;
    float v = (zprev[(row0 + m) * CIN + k] - mean[k]) * rstd[k] * g[k] + be[k];
    Alds[t] = (_Float16)fmaxf(v, 0.f);
  }
  __syncthreads();

  const int lane = tid & 31, wave = tid >> 5;
  const int rt = (ROWT == 2) ? (wave >> 2) : 0;
  const int ct = (ROWT == 2) ? (wave & 3) : wave;
  v8f acc = {};
#pragma unroll
  for (int kt = 0; kt < CIN / 32; ++kt) {
    v16h af = frag_a(Alds + rt * 16 * CIN + kt * 32, CIN, lane);
    v16h bf = frag_b(Wlds + kt * 32 * COUT + ct * 16, COUT, lane);
    acc = __builtin_amdgcn_wmma_f32_16x16x32_f16(false, af, false, bf,
                                                 (short)0, acc, false, false);
  }
  const int  n     = ct * 16 + (lane & 15);
  const float bv   = bias[n];
  const long rbase = row0 + rt * 16 + ((lane < 16) ? 0 : 8);
#pragma unroll
  for (int v = 0; v < 8; ++v) z[(rbase + v) * COUT + n] = acc[v] + bv;
}

// ---------------------------------------------------------------------------
// 5) Deterministic per-channel batch stats: mean + 1/sqrt(var+eps)
// ---------------------------------------------------------------------------
__global__ void k_colstats(const float* __restrict__ z, int Mrows, int C,
                           float* __restrict__ mean, float* __restrict__ rstd) {
  const int c = blockIdx.x;
  float s = 0.f, s2 = 0.f;
  for (long m = threadIdx.x; m < (long)Mrows; m += 256) {
    float v = z[m * C + c]; s += v; s2 += v * v;
  }
  __shared__ float ss[256], ss2[256];
  ss[threadIdx.x] = s; ss2[threadIdx.x] = s2;
  __syncthreads();
  for (int st = 128; st > 0; st >>= 1) {
    if (threadIdx.x < st) {
      ss[threadIdx.x]  += ss[threadIdx.x + st];
      ss2[threadIdx.x] += ss2[threadIdx.x + st];
    }
    __syncthreads();
  }
  if (threadIdx.x == 0) {
    float invM = 1.f / (float)Mrows;
    float mu  = ss[0] * invM;
    float var = ss2[0] * invM - mu * mu;
    mean[c] = mu; rstd[c] = rsqrtf(var + BN_EPS);
  }
}

// ---------------------------------------------------------------------------
// 6) BN + ReLU + max over k neighbors: z[M,C] -> out[B*N, C]
// ---------------------------------------------------------------------------
__global__ void k_maxk(const float* __restrict__ z,
                       const float* __restrict__ mean, const float* __restrict__ rstd,
                       const float* __restrict__ g, const float* __restrict__ be,
                       int C, float* __restrict__ out) {
  long t = (long)blockIdx.x * blockDim.x + threadIdx.x;
  if (t >= (long)BATCH * NPTS * C) return;
  int  c = (int)(t % C);
  long p = t / C;
  const float mu = mean[c], rs = rstd[c], gg = g[c], bb = be[c];
  const float* zp = z + (p * KNN) * C + c;
  float mx = -FINF;
  for (int kk = 0; kk < KNN; ++kk) {
    float v = fmaxf((zp[(long)kk * C] - mu) * rs * gg + bb, 0.f);
    mx = fmaxf(mx, v);
  }
  out[t] = mx;
}

// ---------------------------------------------------------------------------
// 7) squared norms of 64-d features; 8) WMMA Gram -> dist2 matrix
// ---------------------------------------------------------------------------
__global__ void k_sqnorm64(const float* __restrict__ f, float* __restrict__ sq) {
  long p = (long)blockIdx.x * blockDim.x + threadIdx.x;
  if (p >= (long)BATCH * NPTS) return;
  const float* fp = f + p * 64;
  float s = 0.f;
  for (int c = 0; c < 64; ++c) s += fp[c] * fp[c];
  sq[p] = s;
}

__global__ void k_gram64(const float* __restrict__ feat, const float* __restrict__ sq,
                         float* __restrict__ dist) {
  __shared__ _Float16 Ai[16 * 64];        // rows i, [m][k]
  __shared__ _Float16 Bj[8 * 64 * 16];    // per-wave [k][n] = feat[j0+n][k]
  const int tid = threadIdx.x;
  int rem = blockIdx.x;
  const int jg = rem & 15;  rem >>= 4;    // 16 j-groups of 8 tiles
  const int it = rem & 127; rem >>= 7;    // 128 i-tiles
  const int b  = rem;                     // batch
  const int i0 = it * 16;

  for (int t = tid; t < 16 * 64; t += 256) {
    int m = t >> 6, k = t & 63;
    Ai[t] = (_Float16)feat[((long)b * NPTS + i0 + m) * 64 + k];
  }
  for (int t = tid; t < 8 * 64 * 16; t += 256) {
    int w = t >> 10, k = (t >> 4) & 63, n = t & 15;
    int j = (jg * 8 + w) * 16 + n;
    Bj[t] = (_Float16)feat[((long)b * NPTS + j) * 64 + k];
  }
  __syncthreads();

  const int lane = tid & 31, wave = tid >> 5;
  v8f acc = {};
#pragma unroll
  for (int kt = 0; kt < 2; ++kt) {
    v16h af = frag_a(Ai + kt * 32, 64, lane);
    v16h bf = frag_b(Bj + wave * 64 * 16 + kt * 32 * 16, 16, lane);
    acc = __builtin_amdgcn_wmma_f32_16x16x32_f16(false, af, false, bf,
                                                 (short)0, acc, false, false);
  }
  const int j   = (jg * 8 + wave) * 16 + (lane & 15);
  const float sj = sq[(long)b * NPTS + j];
  const int moff = (lane < 16) ? 0 : 8;
#pragma unroll
  for (int v = 0; v < 8; ++v) {
    int i = i0 + v + moff;
    dist[((long)b * NPTS + i) * NPTS + j] = sq[(long)b * NPTS + i] + sj - 2.f * acc[v];
  }
}

// ---------------------------------------------------------------------------
// 9) global max over N; head MLP (tiny, scalar)
// ---------------------------------------------------------------------------
__global__ void k_nmax(const float* __restrict__ x2, float* __restrict__ xg) {
  int t = blockIdx.x * blockDim.x + threadIdx.x;   // 8*128
  if (t >= BATCH * 128) return;
  int b = t / 128, c = t % 128;
  float mx = -FINF;
  for (int n = 0; n < NPTS; ++n)
    mx = fmaxf(mx, x2[((long)b * NPTS + n) * 128 + c]);
  xg[t] = mx;
}

__global__ void k_head1(const float* __restrict__ xg, const float* __restrict__ W5,
                        const float* __restrict__ b5, float* __restrict__ z5) {
  int t = blockIdx.x * blockDim.x + threadIdx.x;   // 8*1024
  if (t >= BATCH * 1024) return;
  int b = t / 1024, c = t % 1024;
  float s = b5[c];
  for (int k = 0; k < 128; ++k) s += xg[b * 128 + k] * W5[k * 1024 + c];
  z5[t] = s;
}

__global__ void k_headout(const float* __restrict__ z5,
                          const float* __restrict__ mean, const float* __restrict__ rstd,
                          const float* __restrict__ g5, const float* __restrict__ be5,
                          const float* __restrict__ W6, const float* __restrict__ b6,
                          float* __restrict__ out) {
  int t = blockIdx.x * blockDim.x + threadIdx.x;   // 8*512
  if (t >= BATCH * 512) return;
  int b = t / 512, co = t % 512;
  float acc = b6[co];
  for (int c = 0; c < 1024; ++c) {
    float h = (z5[b * 1024 + c] - mean[c]) * rstd[c] * g5[c] + be5[c];
    h = fmaxf(h, 0.f);
    acc += h * W6[c * 512 + co];
  }
  out[t] = acc;
}

// ---------------------------------------------------------------------------
extern "C" void kernel_launch(void* const* d_in, const int* in_sizes, int n_in,
                              void* d_out, int out_size, void* d_ws, size_t ws_size,
                              hipStream_t stream) {
  (void)in_sizes; (void)n_in; (void)out_size; (void)ws_size;
  const float* x   = (const float*)d_in[0];
  const float* W1  = (const float*)d_in[1];  const float* b1  = (const float*)d_in[2];
  const float* g1  = (const float*)d_in[3];  const float* be1 = (const float*)d_in[4];
  const float* W2  = (const float*)d_in[5];  const float* b2  = (const float*)d_in[6];
  const float* g2  = (const float*)d_in[7];  const float* be2 = (const float*)d_in[8];
  const float* W3  = (const float*)d_in[9];  const float* b3  = (const float*)d_in[10];
  const float* g3  = (const float*)d_in[11]; const float* be3 = (const float*)d_in[12];
  const float* W4  = (const float*)d_in[13]; const float* b4  = (const float*)d_in[14];
  const float* g4  = (const float*)d_in[15]; const float* be4 = (const float*)d_in[16];
  const float* W5  = (const float*)d_in[17]; const float* b5  = (const float*)d_in[18];
  const float* g5  = (const float*)d_in[19]; const float* be5 = (const float*)d_in[20];
  const float* W6  = (const float*)d_in[21]; const float* b6  = (const float*)d_in[22];
  float* out = (float*)d_out;

  // workspace carve-up (R1 doubles as dist matrix and z buffer; 134MB < 168MB)
  char* w = (char*)d_ws;
  float* R1    = (float*)w; w += (size_t)MEDGE * 128 * 4;   // dist / z1 / z3
  float* R2    = (float*)w; w += (size_t)MEDGE * 128 * 4;   // z2 / z4
  int*   idx   = (int*)  w; w += (size_t)MEDGE * 4;
  float* x1    = (float*)w; w += (size_t)BATCH * NPTS * 64 * 4;
  float* x2    = (float*)w; w += (size_t)BATCH * NPTS * 128 * 4;
  float* sq    = (float*)w; w += (size_t)BATCH * NPTS * 4;
  float* meanb = (float*)w; w += 1024 * 4;
  float* rstdb = (float*)w; w += 1024 * 4;
  float* xg    = (float*)w; w += BATCH * 128 * 4;
  float* z5    = (float*)w; w += BATCH * 1024 * 4;

  // ---- EdgeConv block 1 (D=3) ----
  k_dist1<<<(BATCH * NPTS * NPTS) / 256, 256, 0, stream>>>(x, R1);
  k_select<<<BATCH * NPTS, 128, 0, stream>>>(R1, idx);
  k_edge_gemm<6, 64><<<MEDGE / 32, 256, 0, stream>>>(x, idx, W1, b1, R1);
  k_colstats<<<64, 256, 0, stream>>>(R1, MEDGE, 64, meanb, rstdb);
  k_act_gemm<64, 64><<<MEDGE / 32, 256, 0, stream>>>(R1, meanb, rstdb, g1, be1, W2, b2, R2);
  k_colstats<<<64, 256, 0, stream>>>(R2, MEDGE, 64, meanb, rstdb);
  k_maxk<<<(BATCH * NPTS * 64) / 256, 256, 0, stream>>>(R2, meanb, rstdb, g2, be2, 64, x1);

  // ---- EdgeConv block 2 (D=64) ----
  k_sqnorm64<<<(BATCH * NPTS) / 256, 256, 0, stream>>>(x1, sq);
  k_gram64<<<BATCH * 128 * 16, 256, 0, stream>>>(x1, sq, R1);
  k_select<<<BATCH * NPTS, 128, 0, stream>>>(R1, idx);
  k_edge_gemm<128, 128><<<MEDGE / 16, 256, 0, stream>>>(x1, idx, W3, b3, R1);
  k_colstats<<<128, 256, 0, stream>>>(R1, MEDGE, 128, meanb, rstdb);
  k_act_gemm<128, 128><<<MEDGE / 16, 256, 0, stream>>>(R1, meanb, rstdb, g3, be3, W4, b4, R2);
  k_colstats<<<128, 256, 0, stream>>>(R2, MEDGE, 128, meanb, rstdb);
  k_maxk<<<(BATCH * NPTS * 128) / 256, 256, 0, stream>>>(R2, meanb, rstdb, g4, be4, 128, x2);

  // ---- global pool + head ----
  k_nmax<<<4, 256, 0, stream>>>(x2, xg);
  k_head1<<<32, 256, 0, stream>>>(xg, W5, b5, z5);
  k_colstats<<<1024, 256, 0, stream>>>(z5, BATCH, 1024, meanb, rstdb);
  k_headout<<<16, 256, 0, stream>>>(z5, meanb, rstdb, g5, be5, W6, b6, out);
}
